// GNN_79482664779988
// MI455X (gfx1250) — matrix operand order repaired and checked by reading, not compiled, not run on hardware
//
#include <hip/hip_runtime.h>
#include <hip/hip_bf16.h>

#define N_NODES  50000
#define N_EDGES  200000
#define NODE_DIM 74
#define EDGE_DIM 13
#define FIN      87      // NODE_DIM + EDGE_DIM
#define FINP     96      // concat features padded to 3*32
#define HID      200
#define KPAD     224     // hidden padded to 7*32 (row stride everywhere)
#define NGRAPHS  2048
#define NEGSLOPE 0.01f
#define BNEPS    1e-5f

typedef __attribute__((ext_vector_type(16))) __bf16 v16bf;
typedef __attribute__((ext_vector_type(8)))  float  v8f;

// ---------------------------------------------------------------------------
// CDNA5 WMMA fragment layouts (ISA 7.12.2, wave32), bf16 16x16x32:
//  A: lane = M | khalf<<4. Per 32-wide K-chunk the lane's 16 elements are one
//     run of 16 consecutive K: [kb*32+khalf*16, +16) (this code's order, same as B).
//  B: lane = N | khalf<<4; element t holds K = kb*32 + khalf*16 + t (this code's order) -> the
//     fragment is 32 CONTIGUOUS bytes of a bf16 row (K-major).
//  C/D: lane = N | khalf<<4; VGPR v holds row M = v + 8*khalf.
// ---------------------------------------------------------------------------

// Build A fragment from one padded fp32 row (16 contiguous floats, 4x b128).
__device__ __forceinline__ v16bf frag_a(const float4* __restrict__ r4,
                                        int kb, int khalf)
{
    int o1 = (kb * 32 + khalf * 16) >> 2;
    int o2 = o1 + 2;
    float4 p0 = r4[o1], p1 = r4[o1 + 1], p2 = r4[o2], p3 = r4[o2 + 1];
    v16bf a;
    a[0] = (__bf16)p0.x;  a[1] = (__bf16)p0.y;  a[2]  = (__bf16)p0.z;  a[3]  = (__bf16)p0.w;
    a[4] = (__bf16)p1.x;  a[5] = (__bf16)p1.y;  a[6]  = (__bf16)p1.z;  a[7]  = (__bf16)p1.w;
    a[8] = (__bf16)p2.x;  a[9] = (__bf16)p2.y;  a[10] = (__bf16)p2.z;  a[11] = (__bf16)p2.w;
    a[12] = (__bf16)p3.x; a[13] = (__bf16)p3.y; a[14] = (__bf16)p3.z;  a[15] = (__bf16)p3.w;
    return a;
}

// Build A fragment from the difference of two padded fp32 rows.
__device__ __forceinline__ v16bf frag_a_diff(const float4* __restrict__ u4,
                                             const float4* __restrict__ w4,
                                             int kb, int khalf)
{
    int o1 = (kb * 32 + khalf * 16) >> 2;
    int o2 = o1 + 2;
    float4 p0 = u4[o1], p1 = u4[o1 + 1], p2 = u4[o2], p3 = u4[o2 + 1];
    float4 q0 = w4[o1], q1 = w4[o1 + 1], q2 = w4[o2], q3 = w4[o2 + 1];
    v16bf a;
    a[0]  = (__bf16)(p0.x - q0.x); a[1]  = (__bf16)(p0.y - q0.y);
    a[2]  = (__bf16)(p0.z - q0.z); a[3]  = (__bf16)(p0.w - q0.w);
    a[4]  = (__bf16)(p1.x - q1.x); a[5]  = (__bf16)(p1.y - q1.y);
    a[6]  = (__bf16)(p1.z - q1.z); a[7]  = (__bf16)(p1.w - q1.w);
    a[8]  = (__bf16)(p2.x - q2.x); a[9]  = (__bf16)(p2.y - q2.y);
    a[10] = (__bf16)(p2.z - q2.z); a[11] = (__bf16)(p2.w - q2.w);
    a[12] = (__bf16)(p3.x - q3.x); a[13] = (__bf16)(p3.y - q3.y);
    a[14] = (__bf16)(p3.z - q3.z); a[15] = (__bf16)(p3.w - q3.w);
    return a;
}

// B fragment: 32 contiguous bytes of a bf16 weight row.
__device__ __forceinline__ v16bf frag_b(const __bf16* __restrict__ wrow,
                                        int kb, int khalf)
{
    return *(const v16bf*)(wrow + kb * 32 + khalf * 16);
}

// ---- weight prep: bf16 conversion (K-padded with zeros) + BN folding ------
__global__ void __launch_bounds__(256)
k_prep(const float* __restrict__ Wc_f, const float* __restrict__ Wf_f,
       const float* __restrict__ Wei_f, const float* __restrict__ cb,
       const float* __restrict__ gamma, const float* __restrict__ beta,
       const float* __restrict__ mean, const float* __restrict__ var,
       __bf16* __restrict__ Wc, __bf16* __restrict__ Wf,
       __bf16* __restrict__ Wei, float* __restrict__ alpha,
       float* __restrict__ bias2)
{
    int tid = blockIdx.x * blockDim.x + threadIdx.x;
    if (tid >= HID * KPAD) return;
    int r = tid / KPAD, k = tid - r * KPAD;
    Wc[tid] = (__bf16)(k < HID ? Wc_f[r * HID + k] : 0.f);
    Wf[tid] = (__bf16)(k < HID ? Wf_f[r * HID + k] : 0.f);
    if (k < FINP) Wei[r * FINP + k] = (__bf16)(k < FIN ? Wei_f[r * FIN + k] : 0.f);
    if (k == 0) {
        float sc = gamma[r] * rsqrtf(var[r] + BNEPS);
        alpha[r] = sc;
        bias2[r] = (cb[r] - mean[r]) * sc + beta[r];
    }
}

// ---- generic zero ---------------------------------------------------------
__global__ void __launch_bounds__(256)
k_zero(float* __restrict__ p, long long n)
{
    long long tid = (long long)blockIdx.x * blockDim.x + threadIdx.x;
    if (tid < n) p[tid] = 0.f;
}

// ---- gather+concat: xc[e][0:96] = [x[row[e]] || edge_attr[e] || 0pad] -----
__global__ void __launch_bounds__(256)
k_concat(const float* __restrict__ x, const float* __restrict__ eattr,
         const int* __restrict__ row, float* __restrict__ xc)
{
    long long tid = (long long)blockIdx.x * blockDim.x + threadIdx.x;
    if (tid >= (long long)N_EDGES * FINP) return;
    int e = (int)(tid / FINP);
    int k = (int)(tid - (long long)e * FINP);
    float v = 0.f;
    if (k < NODE_DIM)      v = x[(size_t)row[e] * NODE_DIM + k];
    else if (k < FIN)      v = eattr[(size_t)e * EDGE_DIM + (k - NODE_DIM)];
    xc[tid] = v;
}

// ---- edge init: e = leaky_relu(xc @ Wei^T) --------------------------------
__global__ void __launch_bounds__(256)
k_edge_init(const float* __restrict__ xc, const __bf16* __restrict__ Wei,
            float* __restrict__ e_out)
{
    int wave = (blockIdx.x * blockDim.x + threadIdx.x) >> 5;
    int lane = threadIdx.x & 31;
    if (wave >= N_EDGES / 16) return;
    int mrow = lane & 15, khalf = lane >> 4;
    int edge = wave * 16 + mrow;
    const float4* x4 = (const float4*)(xc + (size_t)edge * FINP);

    v16bf afrag[3];
#pragma unroll
    for (int kb = 0; kb < 3; kb++) afrag[kb] = frag_a(x4, kb, khalf);

    int nc = lane & 15;
    for (int nt = 0; nt < 13; nt++) {
        int n = nt * 16 + nc;
        const __bf16* wr = Wei + (size_t)(n < HID ? n : 0) * FINP;
        v8f acc = {};
#pragma unroll
        for (int kb = 0; kb < 3; kb++) {
            acc = __builtin_amdgcn_wmma_f32_16x16x32_bf16(
                false, afrag[kb], false, frag_b(wr, kb, khalf),
                (short)0, acc, false, false);
        }
        if (n < HID) {
            float* dst = e_out + (size_t)(wave * 16 + 8 * khalf) * KPAD + n;
#pragma unroll
            for (int v = 0; v < 8; v++) {
                float val = acc[v];
                dst[(size_t)v * KPAD] = (val >= 0.f) ? val : NEGSLOPE * val;
            }
        }
    }
}

// ---- scatter-add (float4 granularity): dst[idx[e]][h..h+3] += src[e][..] --
__global__ void __launch_bounds__(256)
k_scatter4(const float* __restrict__ src, const int* __restrict__ idx,
           float* __restrict__ dst)
{
    long long tid = (long long)blockIdx.x * blockDim.x + threadIdx.x;
    if (tid >= (long long)N_EDGES * (HID / 4)) return;
    int e = (int)(tid / (HID / 4));
    int h = (int)(tid - (long long)e * (HID / 4)) * 4;
    float4 v = *(const float4*)(src + (size_t)e * KPAD + h);
    float* d = dst + (size_t)idx[e] * KPAD + h;
    atomicAdd(d + 0, v.x);
    atomicAdd(d + 1, v.y);
    atomicAdd(d + 2, v.z);
    atomicAdd(d + 3, v.w);
}

// ---- conv layer (in place): e = 2*relu((a_msg[row]-e[edge^1]) @ Wc^T *a+b2)
__global__ void __launch_bounds__(256)
k_conv(const float* __restrict__ amsg, const int* __restrict__ row,
       const __bf16* __restrict__ Wc, const float* __restrict__ alpha,
       const float* __restrict__ bias2, float* __restrict__ e_buf)
{
    int wave = (blockIdx.x * blockDim.x + threadIdx.x) >> 5;
    int lane = threadIdx.x & 31;
    if (wave >= N_EDGES / 16) return;
    int mrow = lane & 15, khalf = lane >> 4;
    int edge = wave * 16 + mrow;
    const float4* a4 = (const float4*)(amsg + (size_t)row[edge] * KPAD);
    const float4* r4 = (const float4*)(e_buf + (size_t)(edge ^ 1) * KPAD);

    // All reverse-pair reads complete here, before any store (pairs tile-local).
    v16bf afrag[7];
#pragma unroll
    for (int kb = 0; kb < 7; kb++) afrag[kb] = frag_a_diff(a4, r4, kb, khalf);

    int nc = lane & 15;
    for (int nt = 0; nt < 13; nt++) {
        int n = nt * 16 + nc;
        const __bf16* wr = Wc + (size_t)(n < HID ? n : 0) * KPAD;
        v8f acc = {};
#pragma unroll
        for (int kb = 0; kb < 7; kb++) {
            acc = __builtin_amdgcn_wmma_f32_16x16x32_bf16(
                false, afrag[kb], false, frag_b(wr, kb, khalf),
                (short)0, acc, false, false);
        }
        if (n < HID) {
            float al = alpha[n], bi = bias2[n];
            float* dst = e_buf + (size_t)(wave * 16 + 8 * khalf) * KPAD + n;
#pragma unroll
            for (int v = 0; v < 8; v++) {
                float eh = fmaxf(acc[v] * al + bi, 0.f);  // relu(BN(lin))
                // leaky_relu(eh) == eh for eh >= 0  ->  residual = 2*eh
                dst[(size_t)v * KPAD] = eh + eh;
            }
        }
    }
}

// ---- ffn GEMM fused with graph mean-pool numerator ------------------------
__global__ void __launch_bounds__(256)
k_ffn_pool(const float* __restrict__ node_h, const __bf16* __restrict__ Wf,
           const int* __restrict__ batch, float* __restrict__ sums)
{
    int wave = (blockIdx.x * blockDim.x + threadIdx.x) >> 5;
    int lane = threadIdx.x & 31;
    if (wave >= N_NODES / 16) return;
    int mrow = lane & 15, khalf = lane >> 4;
    const float4* n4 = (const float4*)(node_h + (size_t)(wave * 16 + mrow) * KPAD);

    v16bf afrag[7];
#pragma unroll
    for (int kb = 0; kb < 7; kb++) afrag[kb] = frag_a(n4, kb, khalf);

    int nc = lane & 15;
    for (int nt = 0; nt < 13; nt++) {
        int n = nt * 16 + nc;
        const __bf16* wr = Wf + (size_t)(n < HID ? n : 0) * KPAD;
        v8f acc = {};
#pragma unroll
        for (int kb = 0; kb < 7; kb++) {
            acc = __builtin_amdgcn_wmma_f32_16x16x32_bf16(
                false, afrag[kb], false, frag_b(wr, kb, khalf),
                (short)0, acc, false, false);
        }
        if (n < HID) {
#pragma unroll
            for (int v = 0; v < 8; v++) {
                int node = wave * 16 + v + 8 * khalf;
                atomicAdd(&sums[(size_t)batch[node] * HID + n], acc[v]);
            }
        }
    }
}

// ---- per-graph node counts ------------------------------------------------
__global__ void __launch_bounds__(256)
k_counts(const int* __restrict__ batch, float* __restrict__ counts)
{
    int tid = blockIdx.x * blockDim.x + threadIdx.x;
    if (tid < N_NODES) atomicAdd(&counts[batch[tid]], 1.f);
}

// ---- final divide into d_out ---------------------------------------------
__global__ void __launch_bounds__(256)
k_final(const float* __restrict__ sums, const float* __restrict__ counts,
        float* __restrict__ out)
{
    int tid = blockIdx.x * blockDim.x + threadIdx.x;
    if (tid >= NGRAPHS * HID) return;
    int g = tid / HID;
    out[tid] = sums[tid] / fmaxf(counts[g], 1.f);
}

extern "C" void kernel_launch(void* const* d_in, const int* in_sizes, int n_in,
                              void* d_out, int out_size, void* d_ws, size_t ws_size,
                              hipStream_t stream) {
    (void)in_sizes; (void)n_in; (void)out_size; (void)ws_size;
    const float* x     = (const float*)d_in[0];
    const float* eat   = (const float*)d_in[1];
    const int*   ei    = (const int*)d_in[2];     // [2, E]
    const int*   batch = (const int*)d_in[3];
    const float* Wei_f = (const float*)d_in[4];
    const float* Wc_f  = (const float*)d_in[5];
    const float* cb    = (const float*)d_in[6];
    const float* gam   = (const float*)d_in[7];
    const float* bet   = (const float*)d_in[8];
    const float* mea   = (const float*)d_in[9];
    const float* var   = (const float*)d_in[10];
    const float* Wf_f  = (const float*)d_in[11];
    const int* row = ei;
    const int* col = ei + N_EDGES;

    // workspace layout (every slice a multiple of 256 B)
    char* w = (char*)d_ws;
    float* e_buf  = (float*)w; w += (size_t)N_EDGES * KPAD * 4;   // 179.2 MB (padded)
    float* xc     = (float*)w; w += (size_t)N_EDGES * FINP * 4;   // 76.8 MB
    float* a_msg  = (float*)w; w += (size_t)N_NODES * KPAD * 4;   // 44.8 MB (also node_h)
    float* sums   = (float*)w; w += (size_t)NGRAPHS * HID * 4;    // 1.6 MB
    float* counts = (float*)w; w += 8192;
    float* alpha  = (float*)w; w += 1024;
    float* bias2  = (float*)w; w += 1024;
    __bf16* Wc    = (__bf16*)w; w += (size_t)HID * KPAD * 2;
    __bf16* Wf    = (__bf16*)w; w += (size_t)HID * KPAD * 2;
    __bf16* Wei   = (__bf16*)w; w += (size_t)HID * FINP * 2;

    const int TB = 256;
    long long nEHP = (long long)N_EDGES * KPAD;   // padded edge-state elems
    long long nNHP = (long long)N_NODES * KPAD;   // padded node-state elems
    long long nXC  = (long long)N_EDGES * FINP;
    long long nSC4 = (long long)N_EDGES * (HID / 4);

    // 1) weights -> bf16 (zero K-pad), fold BN; build concat features;
    //    zero e_buf so its K-pad columns are 0 (conv never touches them).
    k_prep<<<(HID * KPAD + TB - 1) / TB, TB, 0, stream>>>(
        Wc_f, Wf_f, Wei_f, cb, gam, bet, mea, var, Wc, Wf, Wei, alpha, bias2);
    k_concat<<<(int)((nXC + TB - 1) / TB), TB, 0, stream>>>(x, eat, row, xc);
    k_zero<<<(int)((nEHP + TB - 1) / TB), TB, 0, stream>>>(e_buf, nEHP);

    // 2) edge init (12500 tiles, 8 waves/block)
    int tilesE = N_EDGES / 16;
    k_edge_init<<<(tilesE + 7) / 8, TB, 0, stream>>>(xc, Wei, e_buf);

    // 3) DEPTH conv layers (shared weights), in-place edge state
    for (int l = 0; l < 4; l++) {
        k_zero<<<(int)((nNHP + TB - 1) / TB), TB, 0, stream>>>(a_msg, nNHP);
        k_scatter4<<<(int)((nSC4 + TB - 1) / TB), TB, 0, stream>>>(e_buf, col, a_msg);
        k_conv<<<(tilesE + 7) / 8, TB, 0, stream>>>(a_msg, row, Wc, alpha, bias2, e_buf);
    }

    // 4) edge -> node aggregation (reuse a_msg as node_h)
    k_zero<<<(int)((nNHP + TB - 1) / TB), TB, 0, stream>>>(a_msg, nNHP);
    k_scatter4<<<(int)((nSC4 + TB - 1) / TB), TB, 0, stream>>>(e_buf, col, a_msg);

    // 5) ffn + mean-pool
    long long nGH = (long long)NGRAPHS * HID;
    k_zero<<<(int)((nGH + TB - 1) / TB), TB, 0, stream>>>(sums, nGH);
    k_zero<<<(NGRAPHS + TB - 1) / TB, TB, 0, stream>>>(counts, NGRAPHS);
    int tilesN = N_NODES / 16;
    k_ffn_pool<<<(tilesN + 7) / 8, TB, 0, stream>>>(a_msg, Wf, batch, sums);
    k_counts<<<(N_NODES + TB - 1) / TB, TB, 0, stream>>>(batch, counts);
    k_final<<<(NGRAPHS * HID + TB - 1) / TB, TB, 0, stream>>>(sums, counts, (float*)d_out);
}
